// MHSA_46926812676744
// MI455X (gfx1250) — matrix operand order repaired and checked by reading, not compiled, not run on hardware
//
#include <hip/hip_runtime.h>
#include <hip/hip_bf16.h>

// ---------------------------------------------------------------------------
// MHSA for MI455X (gfx1250, wave32, WMMA, TDM).
//  B=2, N=2048, C=1024, H=16, Dh=64.  v == k (reference quirk), scale = C^-0.5.
// Pipeline:
//   1) cvt x -> bf16 ; LDS-tiled cvt+transpose Wq/Wk -> bf16 W^T
//   2) proj GEMM (WMMA bf16, TDM tile fills): Q=(xWq+bq)*scale, K=xWk+bk
//   3) flash attention (WMMA bf16, transposed-score layout trick)
// ---------------------------------------------------------------------------

typedef __attribute__((ext_vector_type(16))) __bf16 v16bf;
typedef __attribute__((ext_vector_type(8)))  __bf16 v8bf;
typedef __attribute__((ext_vector_type(4)))  __bf16 v4bf;
typedef __attribute__((ext_vector_type(8)))  float  v8f;
typedef __attribute__((ext_vector_type(4)))  unsigned int v4u;
typedef __attribute__((ext_vector_type(8)))  int v8i;
typedef __attribute__((ext_vector_type(4)))  int v4i;

#define BDIM  2
#define NTOK  2048
#define CDIM  1024
#define HEADS 16
#define DH    64
#define MROWS (BDIM * NTOK) /* 4096 */

#ifndef __has_builtin
#define __has_builtin(x) 0
#endif

// TDM available? (device pass only; host pass parses the manual fallback)
#if defined(__gfx1250__) && __has_builtin(__builtin_amdgcn_tensor_load_to_lds) \
    && __has_builtin(__builtin_amdgcn_s_wait_tensorcnt)
#define USE_TDM 1
#else
#define USE_TDM 0
#endif
// 6-arg builtin ships with therock-10.0 headers; 5-arg with ROCm 7.2.
#if __has_include(<hip/amd_detail/amd_gfx1250_TDM.h>)
#define TDM_ARGS6 1
#else
#define TDM_ARGS6 0
#endif

union V16U { v8bf h[2]; v16bf v; };
__device__ __forceinline__ v16bf combine(v8bf a, v8bf b) {
  V16U u; u.h[0] = a; u.h[1] = b; return u.v;
}
__device__ __forceinline__ v8f vzero8() {
  v8f z;
#pragma unroll
  for (int e = 0; e < 8; ++e) z[e] = 0.f;
  return z;
}

#if USE_TDM
// Issue a 2D bf16 tile load (tile_d0 x tile_d1, dim0 contiguous) into LDS.
// Descriptor bit layout per CDNA5 ISA ch.8 (D# group0/group1).
__device__ __forceinline__ void tdm_load_2d(unsigned lds_addr, const void* gptr,
                                            unsigned tile_d0, unsigned tile_d1,
                                            unsigned tensor_d0,
                                            unsigned tensor_d1,
                                            unsigned stride_d0 /*elems*/) {
  unsigned long long ga = (unsigned long long)(uintptr_t)gptr;
  v4u g0;
  g0[0] = 1u;                                   // count=1, user, no gather
  g0[1] = lds_addr;                             // LDS byte offset
  g0[2] = (unsigned)(ga & 0xFFFFFFFFu);         // global_addr[31:0]
  g0[3] = (unsigned)((ga >> 32) & 0x01FFFFFFu)  // global_addr[56:32]
          | (2u << 30);                         // type=2 ("image")
  v8i g1;
  g1[0] = (int)(1u << 16);                                   // data_size=2B
  g1[1] = (int)((tensor_d0 & 0xFFFFu) << 16);                // dim0[15:0]
  g1[2] = (int)(((tensor_d0 >> 16) & 0xFFFFu) |
                ((tensor_d1 & 0xFFFFu) << 16));              // dim0hi|dim1lo
  g1[3] = (int)(((tensor_d1 >> 16) & 0xFFFFu) |
                ((tile_d0 & 0xFFFFu) << 16));                // dim1hi|tile0
  g1[4] = (int)(tile_d1 & 0xFFFFu);                          // tile1, tile2=0
  g1[5] = (int)stride_d0;                                    // dim0_stride lo
  g1[6] = 0;
  g1[7] = 0;
  v4i z4 = {0, 0, 0, 0};
#if TDM_ARGS6
  v8i z8 = {0, 0, 0, 0, 0, 0, 0, 0};
  __builtin_amdgcn_tensor_load_to_lds(g0, g1, z4, z4, z8, 0);
#else
  __builtin_amdgcn_tensor_load_to_lds(g0, g1, z4, z4, 0);
#endif
}
#endif

// ---------------------------------------------------------------------------
// 1a) f32 -> bf16, 4 elements/thread
__global__ void cvt_f32_bf16(const float* __restrict__ in,
                             __bf16* __restrict__ out, int n4) {
  int i = blockIdx.x * 256 + threadIdx.x;
  if (i < n4) {
    float4 f = ((const float4*)in)[i];
    v4bf o;
    o[0] = (__bf16)f.x; o[1] = (__bf16)f.y;
    o[2] = (__bf16)f.z; o[3] = (__bf16)f.w;
    ((v4bf*)out)[i] = o;
  }
}

// 1b) W [K][N] f32 -> W^T [N][K] bf16, LDS-tiled (coalesced both directions)
__global__ __launch_bounds__(256) void transpose_cvt(
    const float* __restrict__ W, __bf16* __restrict__ Wt) {
  __shared__ float tile[32][33];
  const int tx = threadIdx.x & 31, ty = threadIdx.x >> 5; // 32 x 8
  const int bn = blockIdx.x * 32, bk = blockIdx.y * 32;
#pragma unroll
  for (int j = 0; j < 4; ++j)
    tile[ty + j * 8][tx] = W[(size_t)(bk + ty + j * 8) * CDIM + bn + tx];
  __syncthreads();
#pragma unroll
  for (int j = 0; j < 4; ++j)
    Wt[(size_t)(bn + ty + j * 8) * CDIM + bk + tx] =
        (__bf16)tile[tx][ty + j * 8];
}

// ---------------------------------------------------------------------------
// 2) Projection GEMM: Out[b][h][tok][d] = bf16( (A @ W + bias) * scale )
//    A:  [4096][1024] bf16 row-major (x);  Bt: [1024][1024] bf16 = W^T.
// Block 256 thr = 8 waves; tile 128x128, k-step 32, double-buffered LDS.
// Tile fills via Tensor Data Mover (wave 0 issues, TENSORcnt-synced).
// All fragments are preloaded into distinct registers so the 8 WMMAs issue
// back-to-back behind a single s_wait_dscnt.
__global__ __launch_bounds__(256) void proj_wmma(
    const __bf16* __restrict__ A, const __bf16* __restrict__ Bt,
    const float* __restrict__ bias, __bf16* __restrict__ Out, float scale) {
  __shared__ __bf16 sA[2][128 * 32];
  __shared__ __bf16 sB[2][128 * 32];
  const int tid = threadIdx.x;
  const int lane = tid & 31, wave = tid >> 5;
  const int lr = lane & 15, lhi = lane >> 4;
  const int bm = blockIdx.x * 128, bn = blockIdx.y * 128;
  const int wm = (wave & 3) * 32, wn = (wave >> 2) * 64;

  v8f acc[2][4];
#pragma unroll
  for (int i = 0; i < 2; ++i)
#pragma unroll
    for (int j = 0; j < 4; ++j) acc[i][j] = vzero8();

#if USE_TDM
  auto fill = [&](int buf, int kb) {
    if (wave == 0) {
      tdm_load_2d((unsigned)(uintptr_t)&sA[buf][0], &A[(size_t)bm * CDIM + kb],
                  32, 128, CDIM, MROWS, CDIM);
      tdm_load_2d((unsigned)(uintptr_t)&sB[buf][0], &Bt[(size_t)bn * CDIM + kb],
                  32, 128, CDIM, CDIM, CDIM);
    }
  };
  auto fill_wait = [&]() {
    if (wave == 0) __builtin_amdgcn_s_wait_tensorcnt(0);
  };
#else
  auto fill = [&](int buf, int kb) {
#pragma unroll
    for (int i = 0; i < 2; ++i) {
      int li = tid + i * 256;      // 0..511
      int r  = li >> 2;            // row 0..127
      int c  = (li & 3) << 3;      // k-col 0,8,16,24
      *(v8bf*)&sA[buf][r * 32 + c] =
          *(const v8bf*)&A[(size_t)(bm + r) * CDIM + kb + c];
      *(v8bf*)&sB[buf][r * 32 + c] =
          *(const v8bf*)&Bt[(size_t)(bn + r) * CDIM + kb + c];
    }
  };
  auto fill_wait = [&]() {};
#endif

  fill(0, 0);
  fill_wait();
  __syncthreads();
  const int KSTEPS = CDIM / 32;
  for (int t = 0; t < KSTEPS; ++t) {
    int cur = t & 1;
    if (t + 1 < KSTEPS) fill(cur ^ 1, (t + 1) * 32);
    // Preload ALL fragments (distinct registers -> no WAR serialization).
    // A-fragments (16x32 bf16): lo lanes K{0..7,16..23}, hi lanes +8
    v16bf af[2];
#pragma unroll
    for (int mt = 0; mt < 2; ++mt) {
      const __bf16* p = &sA[cur][(wm + mt * 16 + lr) * 32 + (lhi ? 8 : 0)];
      af[mt] = combine(*(const v8bf*)p, *(const v8bf*)(p + 16));
    }
    // B-fragments (32x16): lane n = col, lo lanes K{0..7,16..23}, hi lanes +8
    v16bf bfr[4];
#pragma unroll
    for (int nt = 0; nt < 4; ++nt) {
      const __bf16* q = &sB[cur][(wn + nt * 16 + lr) * 32 + (lhi ? 8 : 0)];
      bfr[nt] = combine(*(const v8bf*)q, *(const v8bf*)(q + 16));
    }
#pragma unroll
    for (int nt = 0; nt < 4; ++nt)
#pragma unroll
      for (int mt = 0; mt < 2; ++mt)
        acc[mt][nt] = __builtin_amdgcn_wmma_f32_16x16x32_bf16(
            false, af[mt], false, bfr[nt], (short)0, acc[mt][nt], false, false);
    fill_wait();
    __syncthreads();
  }
  // Epilogue: bias, scale, head-major bf16 store. D layout: lane=col, reg=row.
#pragma unroll
  for (int nt = 0; nt < 4; ++nt) {
    int c = bn + wn + nt * 16 + lr;
    float bv = bias[c];
    int h = c >> 6, d = c & 63;
#pragma unroll
    for (int mt = 0; mt < 2; ++mt) {
#pragma unroll
      for (int r = 0; r < 8; ++r) {
        int m = bm + wm + mt * 16 + r + (lhi ? 8 : 0);
        int b = m >> 11, tok = m & 2047;
        float v = (acc[mt][nt][r] + bv) * scale;
        Out[(((size_t)(b * HEADS + h)) * NTOK + tok) * DH + d] = (__bf16)v;
      }
    }
  }
}

// ---------------------------------------------------------------------------
// 3) Flash attention, transposed-score layout.
//    Per wave: 32 queries (2 tiles). Key loop in blocks of 32 (double-buffered).
//    T = K * Q^T  (lane = query -> in-lane softmax; D-layout of two stacked
//    16-key tiles == A-fragment layout of P for O = P*V).
__global__ __launch_bounds__(128) void attn_wmma(
    const __bf16* __restrict__ Qb,   // [32][2048][64], pre-scaled by C^-0.5
    const __bf16* __restrict__ Kb,   // [32][2048][64]  (V == K)
    float* __restrict__ Out) {       // [2][2048][1024]
  __shared__ __bf16 sK[2][32 * 64];  // [key][d]   (A-frags of K)
  __shared__ __bf16 sKt[2][64 * 32]; // [d][key]   (B-frags of V)
  const int tid = threadIdx.x;
  const int lane = tid & 31, wave = tid >> 5;
  const int lr = lane & 15, lhi = lane >> 4;
  const int bh = blockIdx.y;
  const __bf16* Qh = Qb + (size_t)bh * NTOK * DH;
  const __bf16* Kh = Kb + (size_t)bh * NTOK * DH;
  const int qbase = blockIdx.x * 128 + wave * 32;

  // Q^T B-fragments, held in registers all kernel: [qtile][d-half]
  v16bf qf[2][2];
#pragma unroll
  for (int qt = 0; qt < 2; ++qt)
#pragma unroll
    for (int ks = 0; ks < 2; ++ks) {
      const __bf16* p =
          Qh + (size_t)(qbase + qt * 16 + lr) * DH + ks * 32 + (lhi ? 8 : 0);
      qf[qt][ks] = combine(*(const v8bf*)p, *(const v8bf*)(p + 16));
    }

  v8f oacc[2][4];
#pragma unroll
  for (int i = 0; i < 2; ++i)
#pragma unroll
    for (int j = 0; j < 4; ++j) oacc[i][j] = vzero8();
  float mrun[2] = {-3.0e38f, -3.0e38f};
  float lrun[2] = {0.f, 0.f};

  // Fill: thread owns (key-pair, d-chunk); packs key pairs so the K^T scatter
  // is 8x ds_store_b32 (V B-frag VGPR = two adjacent keys at fixed d).
  auto fillK = [&](int buf, int key0) {
    int p  = tid >> 3;             // key pair 0..15
    int d0 = (tid & 7) << 3;       // d 0,8,..,56
    int k0 = key0 + p * 2;
    v8bf a = *(const v8bf*)&Kh[(size_t)k0 * DH + d0];
    v8bf b = *(const v8bf*)&Kh[(size_t)(k0 + 1) * DH + d0];
    *(v8bf*)&sK[buf][(p * 2) * 64 + d0] = a;
    *(v8bf*)&sK[buf][(p * 2 + 1) * 64 + d0] = b;
#pragma unroll
    for (int j = 0; j < 8; ++j) {
      union { __bf16 h[2]; unsigned u; } pk;
      pk.h[0] = a[j]; pk.h[1] = b[j];
      *(unsigned*)&sKt[buf][(d0 + j) * 32 + p * 2] = pk.u;
    }
  };

  fillK(0, 0);
  __syncthreads();
  const int ITERS = NTOK / 32;
  for (int t = 0; t < ITERS; ++t) {
    int cur = t & 1;
    if (t + 1 < ITERS) fillK(cur ^ 1, (t + 1) * 32);
    // A-fragments of the 32-key tile: [key-subtile][d-half]
    v16bf kf[2][2];
#pragma unroll
    for (int kt = 0; kt < 2; ++kt)
#pragma unroll
      for (int ks = 0; ks < 2; ++ks) {
        const __bf16* p =
            &sK[cur][(kt * 16 + lr) * 64 + ks * 32 + (lhi ? 8 : 0)];
        kf[kt][ks] = combine(*(const v8bf*)p, *(const v8bf*)(p + 16));
      }
    // V B-fragments (depend only on the key block -> shared by both q-tiles)
    v16bf vf[4];
#pragma unroll
    for (int nt = 0; nt < 4; ++nt) {
      const __bf16* p = &sKt[cur][(nt * 16 + lr) * 32 + (lhi ? 8 : 0)];
      vf[nt] = combine(*(const v8bf*)p, *(const v8bf*)(p + 16));
    }
#pragma unroll
    for (int qt = 0; qt < 2; ++qt) {
      // T = K * Q^T : lane = query, regs = keys (r + 8*lhi within subtile)
      v8f Tt[2];
#pragma unroll
      for (int kt = 0; kt < 2; ++kt) {
        v8f a = vzero8();
#pragma unroll
        for (int ks = 0; ks < 2; ++ks)
          a = __builtin_amdgcn_wmma_f32_16x16x32_bf16(
              false, kf[kt][ks], false, qf[qt][ks], (short)0, a, false, false);
        Tt[kt] = a;
      }
      // online softmax over keys: in-lane + one shfl_xor(16)
      float mx = Tt[0][0];
#pragma unroll
      for (int e = 1; e < 8; ++e) mx = fmaxf(mx, Tt[0][e]);
#pragma unroll
      for (int e = 0; e < 8; ++e) mx = fmaxf(mx, Tt[1][e]);
      mx = fmaxf(mx, __shfl_xor(mx, 16, 32));
      float mnew = fmaxf(mrun[qt], mx);
      float alpha = __expf(mrun[qt] - mnew);
      mrun[qt] = mnew;
      float s = 0.f;
      v16bf pf; // P as 16x32 A-fragment: els 0..7 <- key tile0, 8..15 <- tile1
#pragma unroll
      for (int e = 0; e < 8; ++e) {
        float p0 = __expf(Tt[0][e] - mnew);
        float p1 = __expf(Tt[1][e] - mnew);
        s += p0 + p1;
        pf[e]     = (__bf16)p0;
        pf[e + 8] = (__bf16)p1;
      }
      lrun[qt] = lrun[qt] * alpha + s;
      // rescale O accumulators: factor for reg r is alpha of query (r + 8*lhi)
      float al[8];
#pragma unroll
      for (int r = 0; r < 8; ++r) al[r] = __shfl(alpha, r + (lhi << 3), 32);
#pragma unroll
      for (int nt = 0; nt < 4; ++nt)
#pragma unroll
        for (int r = 0; r < 8; ++r) oacc[qt][nt][r] *= al[r];
      // O += P @ V
#pragma unroll
      for (int nt = 0; nt < 4; ++nt)
        oacc[qt][nt] = __builtin_amdgcn_wmma_f32_16x16x32_bf16(
            false, pf, false, vf[nt], (short)0, oacc[qt][nt], false, false);
    }
    __syncthreads();
  }
  // Epilogue: divide by row-sum, store f32 [B,N,C]
  const int b = bh >> 4, h = bh & 15;
#pragma unroll
  for (int qt = 0; qt < 2; ++qt) {
    float ltot = lrun[qt] + __shfl_xor(lrun[qt], 16, 32);
    float inv = 1.0f / ltot;
    float rinv[8];
#pragma unroll
    for (int r = 0; r < 8; ++r) rinv[r] = __shfl(inv, r + (lhi << 3), 32);
#pragma unroll
    for (int nt = 0; nt < 4; ++nt) {
      int d = nt * 16 + lr;
#pragma unroll
      for (int r = 0; r < 8; ++r) {
        int tok = qbase + qt * 16 + r + (lhi ? 8 : 0);
        Out[((size_t)(b * NTOK + tok)) * CDIM + h * DH + d] =
            oacc[qt][nt][r] * rinv[r];
      }
    }
  }
}

// ---------------------------------------------------------------------------
extern "C" void kernel_launch(void* const* d_in, const int* in_sizes, int n_in,
                              void* d_out, int out_size, void* d_ws,
                              size_t ws_size, hipStream_t stream) {
  (void)in_sizes; (void)n_in; (void)out_size; (void)ws_size;
  const float* x  = (const float*)d_in[0];
  const float* Wq = (const float*)d_in[1];
  const float* bq = (const float*)d_in[2];
  const float* Wk = (const float*)d_in[3];
  const float* bk = (const float*)d_in[4];
  float* out = (float*)d_out;

  const int XE = MROWS * CDIM;      // 4M elems
  const int WE = CDIM * CDIM;       // 1M elems
  __bf16* xb  = (__bf16*)d_ws;      // 8 MB
  __bf16* Wqt = xb + XE;            // 2 MB
  __bf16* Wkt = Wqt + WE;           // 2 MB
  __bf16* Qb  = Wkt + WE;           // 8 MB
  __bf16* Kb  = Qb + XE;            // 8 MB   (total 28 MB of d_ws)

  cvt_f32_bf16<<<XE / 4 / 256, 256, 0, stream>>>(x, xb, XE / 4);
  transpose_cvt<<<dim3(CDIM / 32, CDIM / 32), 256, 0, stream>>>(Wq, Wqt);
  transpose_cvt<<<dim3(CDIM / 32, CDIM / 32), 256, 0, stream>>>(Wk, Wkt);

  const float scale = 1.0f / 32.0f; // C^-0.5, folded into Q
  proj_wmma<<<dim3(MROWS / 128, CDIM / 128), 256, 0, stream>>>(xb, Wqt, bq, Qb,
                                                               scale);
  proj_wmma<<<dim3(MROWS / 128, CDIM / 128), 256, 0, stream>>>(xb, Wkt, bk, Kb,
                                                               1.0f);
  attn_wmma<<<dim3(NTOK / 128, BDIM * HEADS), 128, 0, stream>>>(Qb, Kb, out);
}
